// mixprop_1709396984394
// MI455X (gfx1250) — matrix operand order repaired and checked by reading, not compiled, not run on hardware
//
#include <hip/hip_runtime.h>
#include <hip/hip_bf16.h>

typedef __attribute__((ext_vector_type(16))) __bf16 v16bf;
typedef __attribute__((ext_vector_type(8)))  float  v8f;

static constexpr int B_ = 16, C_ = 32, V_ = 512, L_ = 168;
static constexpr float ALPHA_ = 0.05f;

union FragBF {
    v16bf v;
    uint4 u[2];
    unsigned short s[16];
};

__device__ inline unsigned short f32_to_bf16(float f) {
    union { float f; unsigned u; } x; x.f = f;
    unsigned lsb = (x.u >> 16) & 1u;
    unsigned r = x.u + 0x7fffu + lsb;   // round-to-nearest-even
    return (unsigned short)(r >> 16);
}

__device__ inline float ipowf(float x, int n) {
    float r = 1.f;
    for (int i = 0; i < n; ++i) r *= x;
    return r;
}

#define WMMA_BF16(A, B, C) \
    __builtin_amdgcn_wmma_f32_16x16x32_bf16(false, (A).v, false, (B).v, (short)0, (C), false, false)

// One pipeline stage: A-fragment + two B-fragments (2 l-tiles share one A)
struct Set { FragBF a, b0, b1; };

__device__ __forceinline__ void load_set(Set& s, const unsigned short* arow,
                                         const unsigned short* brow, int kb) {
    // A: row M fixed per lane; K = kb + {0..7,16..23} (lanes<16) / {8..15,24..31}
    const unsigned short* pa = arow + kb;
    s.a.u[0] = *(const uint4*)(pa);
    s.a.u[1] = *(const uint4*)(pa + 16);
    // B: lane holds column n = lane%16; K = kb + {0..7,16..23} (lanes<16) / {8..15,24..31}
    const unsigned short* pb = brow + (size_t)kb * L_;
    #pragma unroll
    for (int i = 0; i < 16; ++i) {
        const size_t ko = (size_t)(i < 8 ? i : i + 8) * L_;
        s.b0.s[i] = pb[ko];
        s.b1.s[i] = pb[ko + 16];
    }
}

__device__ __forceinline__ void compute_set(const Set& s, v8f& acc0, v8f& acc1) {
    acc0 = WMMA_BF16(s.a, s.b0, acc0);
    acc1 = WMMA_BF16(s.a, s.b1, acc1);
}

// ---------------------------------------------------------------------------
// Kernel 0: x (fp32) -> Y0 (bf16), vectorized 8 elements / thread
// ---------------------------------------------------------------------------
__global__ __launch_bounds__(256)
void k_cvt(const float* __restrict__ X, unsigned short* __restrict__ Y0, long n8) {
    long i = (long)blockIdx.x * 256 + threadIdx.x;
    if (i >= n8) return;
    const float4* p = (const float4*)(X + i * 8);
    float4 f0 = p[0], f1 = p[1];
    union { uint4 u; unsigned short s[8]; } o;
    o.s[0] = f32_to_bf16(f0.x); o.s[1] = f32_to_bf16(f0.y);
    o.s[2] = f32_to_bf16(f0.z); o.s[3] = f32_to_bf16(f0.w);
    o.s[4] = f32_to_bf16(f1.x); o.s[5] = f32_to_bf16(f1.y);
    o.s[6] = f32_to_bf16(f1.z); o.s[7] = f32_to_bf16(f1.w);
    ((uint4*)Y0)[i] = o.u;
}

// ---------------------------------------------------------------------------
// Kernel 1: AT[w][v] = (adj[v][w] + (v==w)) / rowsum_v   (bf16, transposed)
// ---------------------------------------------------------------------------
__global__ __launch_bounds__(256)
void k_build_at(const float* __restrict__ adj, unsigned short* __restrict__ AT) {
    __shared__ float red[256];
    const int v = blockIdx.x;
    const float* row = adj + (size_t)v * V_;
    float s = 0.f;
    for (int w = threadIdx.x; w < V_; w += 256)
        s += row[w] + (w == v ? 1.0f : 0.0f);
    red[threadIdx.x] = s;
    __syncthreads();
    for (int off = 128; off > 0; off >>= 1) {
        if (threadIdx.x < off) red[threadIdx.x] += red[threadIdx.x + off];
        __syncthreads();
    }
    const float rinv = 1.0f / red[0];
    for (int w = threadIdx.x; w < V_; w += 256) {
        float val = (row[w] + (w == v ? 1.0f : 0.0f)) * rinv;
        AT[(size_t)w * V_ + v] = f32_to_bf16(val);
    }
}

// ---------------------------------------------------------------------------
// Kernel 2: Yout = AT @ Yin  per (b,c) plane, pure bf16, branch-free K-loop.
// Block: 8 waves x [16 w-rows each] x 32 l-cols (2 tiles/wave share A-frag).
// 4-set rotation, prefetch distance = 2 K-steps: loads for step s+2 are
// issued before the WMMAs of step s, so waits are partial, not full drains.
// ---------------------------------------------------------------------------
__global__ __launch_bounds__(256)
void k_prop(const unsigned short* __restrict__ AT,
            const unsigned short* __restrict__ Yin,
            unsigned short* __restrict__ Yout) {
    const int lane = threadIdx.x & 31;
    const int wv   = threadIdx.x >> 5;
    const int w0   = blockIdx.x * 128 + wv * 16;
    const int l0   = blockIdx.y * 32;
    const size_t plane = (size_t)blockIdx.z * V_ * L_;

    const int m  = lane & 15;
    const int hi = lane >> 4;

    const unsigned short* arow = AT + (size_t)(w0 + m) * V_ + hi * 8;
    const unsigned short* brow = Yin + plane + (size_t)(hi * 8) * L_ + l0 + m;

    v8f acc0 = {0.f,0.f,0.f,0.f,0.f,0.f,0.f,0.f};
    v8f acc1 = {0.f,0.f,0.f,0.f,0.f,0.f,0.f,0.f};

    Set S0, S1, S2, S3;
    load_set(S0, arow, brow, 0);
    load_set(S1, arow, brow, 32);

    int kb = 0;
    #pragma unroll 1
    for (int it = 0; it < 3; ++it) {       // steady state: 4 K-steps / iter
        load_set(S2, arow, brow, kb + 64);  compute_set(S0, acc0, acc1);
        load_set(S3, arow, brow, kb + 96);  compute_set(S1, acc0, acc1);
        load_set(S0, arow, brow, kb + 128); compute_set(S2, acc0, acc1);
        load_set(S1, arow, brow, kb + 160); compute_set(S3, acc0, acc1);
        kb += 128;
    }
    // kb == 384: S0 holds K=384, S1 holds K=416
    load_set(S2, arow, brow, 448); compute_set(S0, acc0, acc1);
    load_set(S3, arow, brow, 480); compute_set(S1, acc0, acc1);
    compute_set(S2, acc0, acc1);
    compute_set(S3, acc0, acc1);

    // D: lane holds column n, rows M = r + hi*8
    const int n = lane & 15;
    #pragma unroll
    for (int r = 0; r < 8; ++r) {
        const int wrow = w0 + r + hi * 8;
        const size_t rb = plane + (size_t)wrow * L_;
        if (l0 + n < L_)      Yout[rb + l0 + n]      = f32_to_bf16(acc0[r]);
        if (l0 + 16 + n < L_) Yout[rb + l0 + 16 + n] = f32_to_bf16(acc1[r]);
    }
}

// ---------------------------------------------------------------------------
// Kernel 3: out[b,o,p] = bias[o] + sum_j Weff_j[o,:] @ Yj[b,:,p]
// Weff folds the alpha-mixing coefficients into the conv weights.
// Block: 8 waves = [32 o x 64 pixels]; K=128 in 4 bf16 WMMA chunks.
// ---------------------------------------------------------------------------
__global__ __launch_bounds__(256)
void k_conv(const unsigned short* __restrict__ Y0,
            const unsigned short* __restrict__ Y1,
            const unsigned short* __restrict__ Y2,
            const unsigned short* __restrict__ Y3,
            const float* __restrict__ W,
            const float* __restrict__ bias,
            float* __restrict__ out) {
    __shared__ unsigned short Wc[32][128];
    __shared__ float bs[32];

    const float A = ALPHA_, IA = 1.0f - ALPHA_;
    // Weff_j[o,c] = sum_{k>=j} coef(k,j) * W[o, 32k+c]
    //   j==0: coef = (k==0 ? 1 : A);   j>=1: coef = IA^j * (k==j ? 1 : A)
    for (int t = threadIdx.x; t < 32 * 128; t += 256) {
        const int o = t >> 7, kk = t & 127, j = kk >> 5, c = kk & 31;
        float sum = 0.f;
        for (int k = j; k < 4; ++k) {
            float coef;
            if (j == 0) coef = (k == 0) ? 1.0f : A;
            else        coef = ipowf(IA, j) * ((k == j) ? 1.0f : A);
            sum += coef * W[o * 128 + k * 32 + c];
        }
        Wc[o][kk] = f32_to_bf16(sum);
    }
    if (threadIdx.x < 32) bs[threadIdx.x] = bias[threadIdx.x];
    __syncthreads();

    const int lane = threadIdx.x & 31;
    const int wv   = threadIdx.x >> 5;
    const int ot   = wv & 1;        // o-tile 0/1
    const int pt   = wv >> 1;       // pixel-tile 0..3
    const size_t P = (size_t)V_ * L_;
    const int b    = blockIdx.y;
    const size_t p0 = (size_t)blockIdx.x * 64 + (size_t)pt * 16;

    const int m  = lane & 15;
    const int hi = lane >> 4;

    const unsigned short* Ys[4] = { Y0, Y1, Y2, Y3 };

    v8f acc = {0.f,0.f,0.f,0.f,0.f,0.f,0.f,0.f};

    #pragma unroll
    for (int j = 0; j < 4; ++j) {
        FragBF fa, fb;
        const unsigned short* ap = &Wc[ot * 16 + m][j * 32 + hi * 8];
        fa.u[0] = *(const uint4*)(ap);
        fa.u[1] = *(const uint4*)(ap + 16);

        // lane holds pixel column n = lane%16; channels c = {0..7,16..23} (lanes<16) / {8..15,24..31}
        const unsigned short* bp = Ys[j] + ((size_t)(b * 32 + hi * 8)) * P + p0 + m;
        #pragma unroll
        for (int i = 0; i < 16; ++i)
            fb.s[i] = bp[(size_t)(i < 8 ? i : i + 8) * P];

        acc = WMMA_BF16(fa, fb, acc);
    }

    const int n = lane & 15;
    #pragma unroll
    for (int r = 0; r < 8; ++r) {
        const int o = ot * 16 + r + hi * 8;
        out[((size_t)b * 32 + o) * P + p0 + n] = acc[r] + bs[o];
    }
}

// ---------------------------------------------------------------------------
extern "C" void kernel_launch(void* const* d_in, const int* in_sizes, int n_in,
                              void* d_out, int out_size, void* d_ws, size_t ws_size,
                              hipStream_t stream) {
    const float* x   = (const float*)d_in[0];
    const float* adj = (const float*)d_in[1];
    const float* W   = (const float*)d_in[2];
    const float* b   = (const float*)d_in[3];
    float* out = (float*)d_out;

    char* ws = (char*)d_ws;
    const size_t at_bytes = (size_t)V_ * V_ * sizeof(unsigned short);
    const size_t y_elems  = (size_t)B_ * C_ * V_ * L_;
    const size_t y_bytes  = ((y_elems * sizeof(unsigned short) + 255) / 256) * 256 + 256; // pad for edge overread

    unsigned short* AT = (unsigned short*)ws;
    unsigned short* Y0 = (unsigned short*)(ws + at_bytes);
    unsigned short* Y1 = (unsigned short*)(ws + at_bytes + 1 * y_bytes);
    unsigned short* Y2 = (unsigned short*)(ws + at_bytes + 2 * y_bytes);
    unsigned short* Y3 = (unsigned short*)(ws + at_bytes + 3 * y_bytes);

    k_build_at<<<V_, 256, 0, stream>>>(adj, AT);

    const long n8 = (long)(y_elems / 8);                       // 5505024
    k_cvt<<<(unsigned)((n8 + 255) / 256), 256, 0, stream>>>(x, Y0, n8);

    dim3 gp(V_ / 128, (L_ + 31) / 32, B_ * C_);                // 4 x 6 x 512
    k_prop<<<gp, 256, 0, stream>>>(AT, Y0, Y1);
    k_prop<<<gp, 256, 0, stream>>>(AT, Y1, Y2);
    k_prop<<<gp, 256, 0, stream>>>(AT, Y2, Y3);

    const size_t P = (size_t)V_ * L_;                          // 86016, divisible by 64
    dim3 gc((unsigned)(P / 64), B_);                           // 1344 x 16
    k_conv<<<gc, 256, 0, stream>>>(Y0, Y1, Y2, Y3, W, b, out);
}
